// MultiheadAttention_6966436954152
// MI455X (gfx1250) — compile-verified, matrix operand order checked by reading, not run on hardware
//
#include <hip/hip_runtime.h>
#include <hip/hip_bf16.h>
#include <math.h>

// ---------------------------------------------------------------------------
// MultiheadAttention forward for MI455X (gfx1250), wave32 + WMMA bf16.
// T=2048, B=2, E=1024, H=16, D=64, SCALING = 1/8.
// Outputs (flat, f32): attn_output [T,B,E] | attn_weights [B,H,T,T] |
//                      k_new [T,B,E] | v_new [T,B,E]
// ---------------------------------------------------------------------------

typedef __bf16 bf16;
typedef __attribute__((ext_vector_type(8)))  bf16  v8bf;
typedef __attribute__((ext_vector_type(16))) bf16  v16bf;
typedef __attribute__((ext_vector_type(8)))  float v8f;

#define WMMA_BF16(a, b, c) \
  __builtin_amdgcn_wmma_f32_16x16x32_bf16(false, (a), false, (b), (short)0, (c), false, false)

__device__ __forceinline__ bf16 f2bf(float f) {
  unsigned u = __builtin_bit_cast(unsigned, f);
  unsigned r = (u + 0x7FFFu + ((u >> 16) & 1u)) >> 16;
  unsigned short s = (unsigned short)r;
  return __builtin_bit_cast(bf16, s);
}

// Load one 16-bf16-per-lane WMMA fragment (global or LDS pointer).
// Per ISA (CDNA5 ISA, 16-bit A 16x32): lane holds row (lane&15);
// elements [0..7] = K[klo..klo+7], elements [8..15] = K[klo+16..klo+23],
// klo = (lane>>4)*8.  `p` must already point at rowbase + kstep + klo.
template <typename P>
__device__ __forceinline__ v16bf load_frag(const P* p) {
  v8bf lo = *reinterpret_cast<const v8bf*>(p);
  v8bf hi = *reinterpret_cast<const v8bf*>(p + 16);
  return __builtin_shufflevector(lo, hi, 0, 1, 2, 3, 4, 5, 6, 7,
                                         8, 9, 10, 11, 12, 13, 14, 15);
}

// ---------------------------------------------------------------------------
// f32 -> bf16 elementwise convert
// ---------------------------------------------------------------------------
__global__ __launch_bounds__(256) void mha_cvt_bf16(const float* __restrict__ in,
                                                    bf16* __restrict__ out, int n) {
  int i = blockIdx.x * 256 + threadIdx.x;
  if (i < n) out[i] = f2bf(in[i]);
}

// ---------------------------------------------------------------------------
// C[M,N] = A[M,K] @ W[N,K]^T + bias, optionally scaled; f32 and/or bf16 out.
// 256 threads = 8 waves; wave = 32x32 tile (2x2 WMMA tiles); block = 128x64.
// Fragments loaded straight from global (L2-resident: A 8MB, W 2MB << 192MB).
// ---------------------------------------------------------------------------
__global__ __launch_bounds__(256)
void mha_gemm_xwt(const bf16* __restrict__ A, const bf16* __restrict__ W,
                  const float* __restrict__ bias, float* __restrict__ Cf,
                  bf16* __restrict__ Cb, float scale, int M, int N, int K) {
  const int lane = threadIdx.x & 31;
  const int wave = threadIdx.x >> 5;
  const int wm = wave >> 1;            // 0..3
  const int wn = wave & 1;             // 0..1
  const int m0 = blockIdx.x * 128 + wm * 32;
  const int n0 = blockIdx.y * 64 + wn * 32;
  const int lr  = lane & 15;
  const int klo = (lane >> 4) * 8;

  const bf16* a0p = A + (size_t)(m0 + lr) * K + klo;
  const bf16* a1p = A + (size_t)(m0 + 16 + lr) * K + klo;
  const bf16* b0p = W + (size_t)(n0 + lr) * K + klo;
  const bf16* b1p = W + (size_t)(n0 + 16 + lr) * K + klo;

  v8f c00 = {}, c01 = {}, c10 = {}, c11 = {};
  for (int k = 0; k < K; k += 32) {
    v16bf a0 = load_frag(a0p + k);
    v16bf a1 = load_frag(a1p + k);
    v16bf b0 = load_frag(b0p + k);
    v16bf b1 = load_frag(b1p + k);
    c00 = WMMA_BF16(a0, b0, c00);
    c01 = WMMA_BF16(a0, b1, c01);
    c10 = WMMA_BF16(a1, b0, c10);
    c11 = WMMA_BF16(a1, b1, c11);
  }

  // C layout: lane l, reg i  ->  m = i + 8*(l>>4), n = l&15
  const int cm = (lane >> 4) * 8;
  const int cn = lr;
#pragma unroll
  for (int tile = 0; tile < 4; ++tile) {
    v8f c = (tile == 0) ? c00 : (tile == 1) ? c01 : (tile == 2) ? c10 : c11;
    int mb = m0 + ((tile >> 1) ? 16 : 0);
    int nb = n0 + ((tile & 1) ? 16 : 0);
    float bv = bias[nb + cn];
#pragma unroll
    for (int i = 0; i < 8; ++i) {
      float v = (c[i] + bv) * scale;
      size_t idx = (size_t)(mb + cm + i) * N + (nb + cn);
      if (Cf) Cf[idx] = v;
      if (Cb) Cb[idx] = f2bf(v);
    }
  }
}

// ---------------------------------------------------------------------------
// Repack V: v_new f32 [T,B,E] -> vt bf16 [B*H, D, S]  (vt[(bh*64+d)*2048+s])
// so the P@V B-fragment reads are contiguous in K (=s).
// ---------------------------------------------------------------------------
__global__ __launch_bounds__(256)
void mha_repack_v(const float* __restrict__ vnew, bf16* __restrict__ vt) {
  size_t i = (size_t)blockIdx.x * 256 + threadIdx.x;   // dest index
  int s  = (int)(i & 2047);
  size_t rd = i >> 11;
  int d  = (int)(rd & 63);
  int bh = (int)(rd >> 6);
  int h = bh & 15, b = bh >> 4;
  vt[i] = f2bf(vnew[((size_t)s * 2 + b) * 1024 + h * 64 + d]);
}

// ---------------------------------------------------------------------------
// Fused attention: scores (Q.K^T + masks) -> 256KB LDS slab -> softmax
// -> attn_weights f32 out (+ in-place bf16 pack) -> P@V -> ctx bf16 out.
// Block = 256 thr (8 waves), covers 32 query rows x full S=2048 for one (b,h).
// ---------------------------------------------------------------------------
__global__ __launch_bounds__(256)
void mha_attn(const bf16* __restrict__ qb, const bf16* __restrict__ kb,
              const bf16* __restrict__ vt, const float* __restrict__ amask,
              const unsigned char* __restrict__ kpm,
              float* __restrict__ wout, bf16* __restrict__ ctx) {
  constexpr int S = 2048, T = 2048, Bc = 2, E = 1024, D = 64;
  __shared__ float sm[32][S];     // 256 KB — CDNA5 320KB LDS/WGP

  const int bh = blockIdx.y;                 // b*16 + h
  const int b = bh >> 4, h = bh & 15;
  const int t0 = blockIdx.x * 32;
  const int lane = threadIdx.x & 31;
  const int wave = threadIdx.x >> 5;
  const int lr  = lane & 15;
  const int klo = (lane >> 4) * 8;
  const int cm = (lane >> 4) * 8;
  const int cn = lr;

  // ---- phase 1: scores = Q.K^T (q pre-scaled) + attn_mask, kpm -> -inf.
  // Tile stride is 8 waves (even), so each wave's M-tile parity is constant:
  // hoist the two loop-invariant Q fragments out of the tile loop.
  const int mt = wave & 1;
  {
    const bf16* qp = qb + ((size_t)(t0 + mt * 16 + lr) * Bc + b) * E + h * D + klo;
    const v16bf q0 = load_frag(qp);
    const v16bf q1 = load_frag(qp + 32);
    for (int tile = wave; tile < 2 * (S / 16); tile += 8) {
      int s0 = (tile >> 1) * 16;
      const bf16* kp = kb + ((size_t)(s0 + lr) * Bc + b) * E + h * D + klo;
      v8f acc = {};
      acc = WMMA_BF16(q0, load_frag(kp),      acc);
      acc = WMMA_BF16(q1, load_frag(kp + 32), acc);
#pragma unroll
      for (int i = 0; i < 8; ++i) {
        int m  = mt * 16 + cm + i;           // row within 32-row slab
        int sg = s0 + cn;
        float v = acc[i] + amask[(size_t)(t0 + m) * T + sg];
        if (kpm[b * S + sg]) v = -__builtin_inff();
        sm[m][sg] = v;
      }
    }
  }
  __syncthreads();

  // ---- phase 2: row softmax in LDS; write normalized weights (f32) to d_out
  // and pack normalized probs as bf16 in place into the low half of each row.
  // In-place pack is safe: per iteration the wave's ds_load issues before its
  // ds_store (same-wave LDS ordering), and any float clobbered by the bf16
  // store at element i has float index i/2 <= already-read indices.
  for (int r = wave * 4; r < wave * 4 + 4; ++r) {
    float mx = -__builtin_inff();
    for (int i = lane; i < S; i += 32) mx = fmaxf(mx, sm[r][i]);
    for (int o = 16; o > 0; o >>= 1) mx = fmaxf(mx, __shfl_xor(mx, o, 32));
    float sum = 0.f;
    for (int i = lane; i < S; i += 32) {
      float e = expf(sm[r][i] - mx);
      sm[r][i] = e;
      sum += e;
    }
    for (int o = 16; o > 0; o >>= 1) sum += __shfl_xor(sum, o, 32);
    float inv = 1.f / sum;
    float* wrow = wout + ((size_t)bh * T + (t0 + r)) * S;
    bf16* smb = (bf16*)&sm[r][0];
    for (int i = lane; i < S; i += 32) {
      float p = sm[r][i] * inv;
      wrow[i] = p;
      smb[i] = f2bf(p);
    }
  }
  __syncthreads();

  // ---- phase 3: ctx = P @ V ; 8 waves = 2 Mtiles x 4 Ntiles, K = S.
  // A-fragments come from the bf16-packed LDS rows as two ds_load_b128s.
  {
    const int pmt = wave & 1;
    const int nt = wave >> 1;                         // 0..3 -> d-cols
    const bf16* ap = (const bf16*)&sm[pmt * 16 + lr][0];
    const bf16* vp = vt + ((size_t)bh * D + nt * 16 + lr) * S + klo;
    v8f acc = {};
    for (int ks = 0; ks < S; ks += 32) {
      v16bf a = load_frag(ap + ks + klo);
      acc = WMMA_BF16(a, load_frag(vp + ks), acc);
    }
#pragma unroll
    for (int i = 0; i < 8; ++i) {
      int tg = t0 + pmt * 16 + cm + i;
      int d  = nt * 16 + cn;
      ctx[((size_t)tg * Bc + b) * E + h * D + d] = f2bf(acc[i]);
    }
  }
}

// ---------------------------------------------------------------------------
extern "C" void kernel_launch(void* const* d_in, const int* in_sizes, int n_in,
                              void* d_out, int out_size, void* d_ws, size_t ws_size,
                              hipStream_t stream) {
  (void)in_sizes; (void)n_in; (void)out_size; (void)ws_size;
  constexpr int T = 2048, B = 2, E = 1024, H = 16;
  constexpr int M = T * B;                 // 4096 rows for projections
  constexpr float SCALING = 0.125f;        // 64^-0.5

  const float* query = (const float*)d_in[0];
  const unsigned char* kpm = (const unsigned char*)d_in[1];
  const float* amask = (const float*)d_in[2];
  const float* Wq = (const float*)d_in[3]; const float* bq = (const float*)d_in[4];
  const float* Wk = (const float*)d_in[5]; const float* bk = (const float*)d_in[6];
  const float* Wv = (const float*)d_in[7]; const float* bv = (const float*)d_in[8];
  const float* Wo = (const float*)d_in[9]; const float* bo = (const float*)d_in[10];

  float* attn_output = (float*)d_out;                       // [T,B,E]
  float* weights     = attn_output + (size_t)T * B * E;     // [B,H,T,T]
  float* k_new       = weights + (size_t)B * H * T * T;     // [T,B,E]
  float* v_new       = k_new + (size_t)T * B * E;           // [T,B,E]

  char* ws = (char*)d_ws;
  const size_t MB = 1ull << 20;
  bf16* x_bf   = (bf16*)(ws + 0 * MB);     // 8MB  query bf16
  bf16* q_bf   = (bf16*)(ws + 8 * MB);     // 8MB  scaled q
  bf16* k_bf   = (bf16*)(ws + 16 * MB);    // 8MB
  bf16* ctx_bf = (bf16*)(ws + 24 * MB);    // 8MB  attention output pre-Wo
  bf16* vt_bf  = (bf16*)(ws + 32 * MB);    // 8MB  V transposed [bh,d,s]
  bf16* Wq_bf  = (bf16*)(ws + 40 * MB);    // 2MB each
  bf16* Wk_bf  = (bf16*)(ws + 42 * MB);
  bf16* Wv_bf  = (bf16*)(ws + 44 * MB);
  bf16* Wo_bf  = (bf16*)(ws + 46 * MB);

  // convert inputs to bf16
  mha_cvt_bf16<<<(M * E) / 256, 256, 0, stream>>>(query, x_bf, M * E);
  mha_cvt_bf16<<<(E * E) / 256, 256, 0, stream>>>(Wq, Wq_bf, E * E);
  mha_cvt_bf16<<<(E * E) / 256, 256, 0, stream>>>(Wk, Wk_bf, E * E);
  mha_cvt_bf16<<<(E * E) / 256, 256, 0, stream>>>(Wv, Wv_bf, E * E);
  mha_cvt_bf16<<<(E * E) / 256, 256, 0, stream>>>(Wo, Wo_bf, E * E);

  dim3 gg(M / 128, E / 64);
  // q = (x@Wq^T + bq) * scaling            (bf16 only, internal)
  mha_gemm_xwt<<<gg, 256, 0, stream>>>(x_bf, Wq_bf, bq, (float*)nullptr, q_bf, SCALING, M, E, E);
  // k_new = x@Wk^T + bk                    (f32 out + bf16 internal)
  mha_gemm_xwt<<<gg, 256, 0, stream>>>(x_bf, Wk_bf, bk, k_new, k_bf, 1.0f, M, E, E);
  // v_new = x@Wv^T + bv                    (f32 out only)
  mha_gemm_xwt<<<gg, 256, 0, stream>>>(x_bf, Wv_bf, bv, v_new, (bf16*)nullptr, 1.0f, M, E, E);

  // V -> [bh, d, s] bf16 for contiguous P@V fragments
  mha_repack_v<<<(B * H * 64 * T) / 256, 256, 0, stream>>>(v_new, vt_bf);

  // fused scores + softmax + P@V
  dim3 ga(T / 32, B * H);
  mha_attn<<<ga, 256, 0, stream>>>(q_bf, k_bf, vt_bf, amask, kpm, weights, ctx_bf);

  // attn_output = ctx@Wo^T + bo
  mha_gemm_xwt<<<gg, 256, 0, stream>>>(ctx_bf, Wo_bf, bo, attn_output, (bf16*)nullptr, 1.0f, M, E, E);
}
